// localTransformer_70540542869683
// MI455X (gfx1250) — compile-verified, matrix operand order checked by reading, not run on hardware
//
#include <hip/hip_runtime.h>
#include <hip/hip_bf16.h>
#include <math.h>

#define BB   4
#define NN   2048
#define KK   16
#define CIN  480
#define DD   64
#define N4   (4 * NN)
#define KN   (KK * NN)
#define EPSV 1e-5f

typedef __attribute__((ext_vector_type(16))) _Float16 v16h;
typedef __attribute__((ext_vector_type(8)))  _Float16 v8h;
typedef __attribute__((ext_vector_type(8)))  float    v8f;

// ---------------------------------------------------------------------------
// WMMA helpers (gfx1250 wave32, v_wmma_f32_16x16x32_f16).
// Fragment layout per CDNA5 ISA 7.12.2 (f16 A/B):
//   lane i<16 -> row/col i, k in [kb,kb+8) U [kb+16,kb+24), kb = (lane>=16)*8
// With k-contiguous storage each fragment = two 16-byte vector loads.
//   C/D (f32): n = lane&15, m = v + (lane<16 ? 0 : 8)
// ---------------------------------------------------------------------------
__device__ __forceinline__ v16h load_frag_k1(const _Float16* p, long si) {
  const int lane = threadIdx.x & 31;
  const int i = lane & 15;
  const int kb = (lane >> 4) * 8;
  const v8h* q = (const v8h*)(p + (long)i * si + kb);
  const v8h lo = q[0];
  const v8h hi = q[2];  // +16 halves
  return __builtin_shufflevector(lo, hi, 0, 1, 2, 3, 4, 5, 6, 7,
                                 8, 9, 10, 11, 12, 13, 14, 15);
}

__device__ __forceinline__ v8f wmma_gemm(v8f acc,
                                         const _Float16* A, long a_si,
                                         const _Float16* Bp, long b_si,
                                         int kdim) {
  for (int k0 = 0; k0 < kdim; k0 += 32) {
    v16h a = load_frag_k1(A + k0, a_si);
    v16h b = load_frag_k1(Bp + k0, b_si);
    acc = __builtin_amdgcn_wmma_f32_16x16x32_f16(false, a, false, b,
                                                 (short)0, acc, false, false);
  }
  return acc;
}

// ---------------------------------------------------------------------------
// f32 -> f16 convert (weights; already k-contiguous row-major (O,K))
// ---------------------------------------------------------------------------
__global__ void cvt_f32_f16_kernel(const float* __restrict__ s,
                                   _Float16* __restrict__ d, int n) {
  for (int i = blockIdx.x * blockDim.x + threadIdx.x; i < n;
       i += gridDim.x * blockDim.x)
    d[i] = (_Float16)s[i];
}

// featT[(b*NN+n)*CIN + c] = feature[(b*CIN+c)*NN + n]  (f16, c-innermost)
__global__ void feat_transpose_kernel(const float* __restrict__ f,
                                      _Float16* __restrict__ ft) {
  const long total = (long)BB * NN * CIN;
  for (long i = (long)blockIdx.x * blockDim.x + threadIdx.x; i < total;
       i += (long)gridDim.x * blockDim.x) {
    const long b = i / ((long)NN * CIN);
    const long r = i % ((long)NN * CIN);
    const long n = r / CIN;
    const long c = r % CIN;
    ft[i] = (_Float16)f[(b * CIN + c) * NN + n];
  }
}

// atT[r][o][c] = at_w[c][o][r] (f16), plus folded BN scales g/sqrt(1+eps)
__global__ void prep_misc_kernel(const float* __restrict__ at_w,
                                 const float* __restrict__ fd_g,
                                 const float* __restrict__ fg_g,
                                 _Float16* __restrict__ atT,
                                 float* __restrict__ gsA,
                                 float* __restrict__ gsG) {
  const int t = threadIdx.x;
  for (int i = t; i < 4 * DD * DD; i += blockDim.x) {
    int r = i / (DD * DD);
    int o = (i / DD) % DD;
    int c = i % DD;
    atT[i] = (_Float16)at_w[(c * DD + o) * 4 + r];
  }
  const float s = rsqrtf(1.0f + EPSV);
  for (int i = t; i < DD; i += blockDim.x)       gsA[i] = fd_g[i] * s;
  for (int i = t; i < 4 * DD; i += blockDim.x)   gsG[i] = fg_g[i] * s;
}

// ---------------------------------------------------------------------------
// KNN: one thread per (b,n), top-(K+1) insertion sort, drop self.
// ---------------------------------------------------------------------------
__global__ void knn_kernel(const float* __restrict__ xyz, int* __restrict__ idx) {
  const int gid = blockIdx.x * blockDim.x + threadIdx.x;
  if (gid >= BB * NN) return;
  const int b = gid / NN, n = gid % NN;
  const float* X = xyz + (long)b * 3 * NN;
  const float px = X[n], py = X[NN + n], pz = X[2 * NN + n];
  float bd[KK + 1];
  int   bi[KK + 1];
#pragma unroll
  for (int j = 0; j <= KK; ++j) { bd[j] = 3.4e38f; bi[j] = 0; }
  for (int m = 0; m < NN; ++m) {
    __builtin_prefetch(X + m + 128, 0, 1);
    const float dx = px - X[m], dy = py - X[NN + m], dz = pz - X[2 * NN + m];
    const float d = dx * dx + dy * dy + dz * dz;
    if (d < bd[KK]) {
      int j = KK;
      while (j > 0 && bd[j - 1] > d) {
        bd[j] = bd[j - 1]; bi[j] = bi[j - 1]; --j;
      }
      bd[j] = d; bi[j] = m;
    }
  }
  for (int j = 0; j < KK; ++j)
    idx[((long)b * KK + j) * NN + n] = bi[j + 1];  // drop self
}

// ---------------------------------------------------------------------------
// Backbone: x = bt2@relu(bt1@F+b1) + bts@F + b ; q,k,v = W@x + b.
// One block = 16 columns of one batch, 4 waves = 4 x 16-row M tiles.
// LDS tiles stored [col][k] so B-fragments are ds_load_b128 pairs.
// ---------------------------------------------------------------------------
__global__ void backbone_kernel(const _Float16* __restrict__ featT,
                                const _Float16* __restrict__ bt1_h, const float* __restrict__ bt1_b,
                                const _Float16* __restrict__ bt2_h, const float* __restrict__ bt2_b,
                                const _Float16* __restrict__ bts_h, const float* __restrict__ bts_b,
                                const _Float16* __restrict__ wq_h, const float* __restrict__ q_b,
                                const _Float16* __restrict__ wk_h, const float* __restrict__ k_b,
                                const _Float16* __restrict__ wv_h, const float* __restrict__ v_b,
                                float* __restrict__ xo, float* __restrict__ qo,
                                float* __restrict__ ko, float* __restrict__ vo) {
  __shared__ __align__(16) _Float16 h_s[16][DD];
  __shared__ __align__(16) _Float16 x_s[16][DD];
  const int b  = blockIdx.x / (NN / 16);
  const int n0 = (blockIdx.x % (NN / 16)) * 16;
  const int w = threadIdx.x >> 5, lane = threadIdx.x & 31;
  const int cn = lane & 15, mb = (lane < 16) ? 0 : 8, m0 = w * 16;
  const _Float16* F = featT + ((long)b * NN + n0) * CIN;  // (n,k): F[n*CIN+k]

  v8f acc = {};
  acc = wmma_gemm(acc, bt1_h + (long)m0 * CIN, CIN, F, CIN, CIN);
#pragma unroll
  for (int e = 0; e < 8; ++e) {
    const int m = m0 + mb + e;
    h_s[cn][m] = (_Float16)fmaxf(acc[e] + bt1_b[m], 0.0f);
  }
  __syncthreads();

  v8f a2 = {};
  a2 = wmma_gemm(a2, bt2_h + (long)m0 * DD, DD, &h_s[0][0], DD, DD);
  a2 = wmma_gemm(a2, bts_h + (long)m0 * CIN, CIN, F, CIN, CIN);
#pragma unroll
  for (int e = 0; e < 8; ++e) {
    const int m = m0 + mb + e;
    const float xv = a2[e] + bt2_b[m] + bts_b[m];
    x_s[cn][m] = (_Float16)xv;
    xo[((long)b * DD + m) * NN + n0 + cn] = xv;
  }
  __syncthreads();

  v8f aq = {}, ak = {}, av = {};
  aq = wmma_gemm(aq, wq_h + (long)m0 * DD, DD, &x_s[0][0], DD, DD);
  ak = wmma_gemm(ak, wk_h + (long)m0 * DD, DD, &x_s[0][0], DD, DD);
  av = wmma_gemm(av, wv_h + (long)m0 * DD, DD, &x_s[0][0], DD, DD);
#pragma unroll
  for (int e = 0; e < 8; ++e) {
    const int m = m0 + mb + e;
    const long o = ((long)b * DD + m) * NN + n0 + cn;
    qo[o] = aq[e] + q_b[m];
    ko[o] = ak[e] + k_b[m];
    vo[o] = av[e] + v_b[m];
  }
}

// ---------------------------------------------------------------------------
// pos_enc + fused gathers: value = v_local + pe ; s = q - k_local + pe.
// s stored k-innermost: sT[(b*KN+col)*64 + c].
// ---------------------------------------------------------------------------
__global__ void posenc_kernel(const float* __restrict__ xyz, const int* __restrict__ idx,
                              const float* __restrict__ fd1_w, const float* __restrict__ fd1_b,
                              const float* __restrict__ gsA, const float* __restrict__ fd_bb,
                              const _Float16* __restrict__ fd2_h, const float* __restrict__ fd2_b,
                              const float* __restrict__ qo, const float* __restrict__ ko,
                              const float* __restrict__ vo,
                              float* __restrict__ value, _Float16* __restrict__ sT) {
  __shared__ float    rel_s[3][16];
  __shared__ int      mi_s[16];
  __shared__ __align__(16) _Float16 t_s[16][DD];
  const int b  = blockIdx.x / (KN / 16);
  const int c0 = (blockIdx.x % (KN / 16)) * 16;
  const int tid = threadIdx.x;

  if (tid < 16) {
    const int col = c0 + tid;
    const int j = col / NN, n = col % NN;
    const int m = idx[((long)b * KK + j) * NN + n];
    mi_s[tid] = m;
    const float* X = xyz + (long)b * 3 * NN;
    rel_s[0][tid] = X[n]          - X[m];
    rel_s[1][tid] = X[NN + n]     - X[NN + m];
    rel_s[2][tid] = X[2 * NN + n] - X[2 * NN + m];
  }
  __syncthreads();

  for (int e = tid; e < 16 * DD; e += blockDim.x) {
    const int cc = e >> 6, o = e & 63;
    float d = fd1_w[o * 3] * rel_s[0][cc] + fd1_w[o * 3 + 1] * rel_s[1][cc] +
              fd1_w[o * 3 + 2] * rel_s[2][cc];
    d = (d + fd1_b[o]) * gsA[o] + fd_bb[o];
    t_s[cc][o] = (_Float16)fmaxf(d, 0.0f);
  }
  __syncthreads();

  const int w = tid >> 5, lane = tid & 31;
  const int cn = lane & 15, mb = (lane < 16) ? 0 : 8, m0 = w * 16;
  v8f acc = {};
  acc = wmma_gemm(acc, fd2_h + (long)m0 * DD, DD, &t_s[0][0], DD, DD);

  const int col = c0 + cn;
  const int n = col % NN;
  const int mg = mi_s[cn];
#pragma unroll
  for (int e = 0; e < 8; ++e) {
    const int m = m0 + mb + e;
    const float pe = acc[e] + fd2_b[m];
    const long  ch = (long)b * DD + m;
    value[ch * KN + col] = pe + vo[ch * NN + mg];         // (b,m,j,n)
    sT[((long)b * KN + col) * DD + m] =
        (_Float16)(qo[ch * NN + n] - ko[ch * NN + mg] + pe);
  }
}

// ---------------------------------------------------------------------------
// attn = fg2 @ relu(bn(fg1 @ s)); hidden tile [16][256] f16 in LDS.
// attn stored fragment-friendly: attnT[((b*NN+n)*KK+j)*64 + c].
// ---------------------------------------------------------------------------
__global__ void attnmlp_kernel(const _Float16* __restrict__ sT,
                               const _Float16* __restrict__ fg1_h, const float* __restrict__ fg1_b,
                               const float* __restrict__ gsG, const float* __restrict__ fg_bb,
                               const _Float16* __restrict__ fg2_h, const float* __restrict__ fg2_b,
                               _Float16* __restrict__ attnT) {
  __shared__ __align__(16) _Float16 hid[16][4 * DD];
  const int b  = blockIdx.x / (KN / 16);
  const int c0 = (blockIdx.x % (KN / 16)) * 16;
  const int w = threadIdx.x >> 5, lane = threadIdx.x & 31;
  const int cn = lane & 15, mb = (lane < 16) ? 0 : 8;
  const _Float16* S = sT + ((long)b * KN + c0) * DD;  // (n,k): S[n*64+k]

  for (int t = 0; t < 4; ++t) {
    const int m0 = w * 64 + t * 16;
    v8f acc = {};
    acc = wmma_gemm(acc, fg1_h + (long)m0 * DD, DD, S, DD, DD);
#pragma unroll
    for (int e = 0; e < 8; ++e) {
      const int m = m0 + mb + e;
      const float hv = (acc[e] + fg1_b[m]) * gsG[m] + fg_bb[m];
      hid[cn][m] = (_Float16)fmaxf(hv, 0.0f);
    }
  }
  __syncthreads();

  const int m0 = w * 16;
  v8f acc = {};
  acc = wmma_gemm(acc, fg2_h + (long)m0 * 256, 256, &hid[0][0], 4 * DD, 4 * DD);
  const int col = c0 + cn;
  const int j = col / NN, n = col % NN;
#pragma unroll
  for (int e = 0; e < 8; ++e) {
    const int m = m0 + mb + e;
    attnT[(((long)b * NN + n) * KK + j) * DD + m] = (_Float16)(acc[e] + fg2_b[m]);
  }
}

// ---------------------------------------------------------------------------
// ConvTranspose(4,1) as 4 per-tap 64x64x16 WMMA GEMMs + channel softmax +
// K-weighted sum with value + x4 identity upsample. One block per (b,n).
// ---------------------------------------------------------------------------
__global__ void fuse_kernel(const _Float16* __restrict__ attnT,
                            const _Float16* __restrict__ atT_h,
                            const float* __restrict__ at_b,
                            const float* __restrict__ value,
                            const float* __restrict__ xo,
                            float* __restrict__ res_out, _Float16* __restrict__ resT) {
  __shared__ float y_s[DD][17];
  __shared__ float cmax[16], csum[16];
  const int b = blockIdx.x / NN;
  const int n = blockIdx.x % NN;
  const int w = threadIdx.x >> 5, lane = threadIdx.x & 31;
  const int cn = lane & 15, mb = (lane < 16) ? 0 : 8, m0 = w * 16;
  const _Float16* A = attnT + ((long)b * NN + n) * KK * DD;  // (f,k): A[f*64+k]

  for (int r = 0; r < 4; ++r) {
    v8f acc = {};
    acc = wmma_gemm(acc, atT_h + ((long)r * DD + m0) * DD, DD, A, DD, DD);
#pragma unroll
    for (int e = 0; e < 8; ++e) {
      const int m = m0 + mb + e;
      y_s[m][cn] = acc[e] + at_b[m];
    }
    __syncthreads();
    if (threadIdx.x < 16) {
      const int f = threadIdx.x;
      float mx = -3.4e38f;
      for (int o = 0; o < DD; ++o) mx = fmaxf(mx, y_s[o][f]);
      float sm = 0.0f;
      for (int o = 0; o < DD; ++o) sm += __expf(y_s[o][f] - mx);
      cmax[f] = mx; csum[f] = sm;
    }
    __syncthreads();
    if (threadIdx.x < DD) {
      const int o = threadIdx.x;
      float accum = 0.0f;
      const float* V = value + ((long)b * DD + o) * KN + n;
      for (int f = 0; f < KK; ++f)
        accum += __expf(y_s[o][f] - cmax[f]) / csum[f] * V[(long)f * NN];
      const float rv = accum + xo[((long)b * DD + o) * NN + n];
      const long nn = 4 * (long)n + r;
      res_out[((long)b * DD + o) * N4 + nn] = rv;
      resT[((long)b * N4 + nn) * DD + o]   = (_Float16)rv;
    }
    __syncthreads();
  }
}

// ---------------------------------------------------------------------------
// completion = m42 @ relu(m41 @ res + b) + b2   (m41 via WMMA, m42 scalar)
// ---------------------------------------------------------------------------
__global__ void completion_kernel(const _Float16* __restrict__ resT,
                                  const _Float16* __restrict__ m41_h,
                                  const float* __restrict__ m41_b,
                                  const float* __restrict__ m42_w,
                                  const float* __restrict__ m42_b,
                                  float* __restrict__ comp) {
  __shared__ __align__(16) _Float16 h_s[16][DD];
  const int b  = blockIdx.x / (N4 / 16);
  const int c0 = (blockIdx.x % (N4 / 16)) * 16;
  const int w = threadIdx.x >> 5, lane = threadIdx.x & 31;
  const int cn = lane & 15, mb = (lane < 16) ? 0 : 8, m0 = w * 16;
  const _Float16* R = resT + ((long)b * N4 + c0) * DD;  // (n,k): R[n*64+k]

  v8f acc = {};
  acc = wmma_gemm(acc, m41_h + (long)m0 * DD, DD, R, DD, DD);
#pragma unroll
  for (int e = 0; e < 8; ++e) {
    const int m = m0 + mb + e;
    h_s[cn][m] = (_Float16)fmaxf(acc[e] + m41_b[m], 0.0f);
  }
  __syncthreads();

  if (threadIdx.x < 48) {
    const int o = threadIdx.x >> 4, cc = threadIdx.x & 15;
    float s = m42_b[o];
    for (int c = 0; c < DD; ++c) s += m42_w[o * DD + c] * (float)h_s[cc][c];
    comp[((long)b * 3 + o) * N4 + c0 + cc] = s;
  }
}

// ---------------------------------------------------------------------------
extern "C" void kernel_launch(void* const* d_in, const int* in_sizes, int n_in,
                              void* d_out, int out_size, void* d_ws, size_t ws_size,
                              hipStream_t stream) {
  const float* feature = (const float*)d_in[0];
  const float* xyz     = (const float*)d_in[1];
  const float* bt1_w = (const float*)d_in[2];  const float* bt1_b = (const float*)d_in[3];
  const float* bt2_w = (const float*)d_in[4];  const float* bt2_b = (const float*)d_in[5];
  const float* bts_w = (const float*)d_in[6];  const float* bts_b = (const float*)d_in[7];
  const float* q_w   = (const float*)d_in[8];  const float* q_b   = (const float*)d_in[9];
  const float* k_w   = (const float*)d_in[10]; const float* k_b   = (const float*)d_in[11];
  const float* v_w   = (const float*)d_in[12]; const float* v_b   = (const float*)d_in[13];
  const float* fd1_w = (const float*)d_in[14]; const float* fd1_b = (const float*)d_in[15];
  const float* fd_g  = (const float*)d_in[16]; const float* fd_bb = (const float*)d_in[17];
  const float* fd2_w = (const float*)d_in[18]; const float* fd2_b = (const float*)d_in[19];
  const float* fg1_w = (const float*)d_in[20]; const float* fg1_b = (const float*)d_in[21];
  const float* fg_g  = (const float*)d_in[22]; const float* fg_bb = (const float*)d_in[23];
  const float* fg2_w = (const float*)d_in[24]; const float* fg2_b = (const float*)d_in[25];
  const float* at_w  = (const float*)d_in[26]; const float* at_b  = (const float*)d_in[27];
  const float* m41_w = (const float*)d_in[28]; const float* m41_b = (const float*)d_in[29];
  const float* m42_w = (const float*)d_in[30]; const float* m42_b = (const float*)d_in[31];

  float* comp    = (float*)d_out;                       // (B,3,4N)
  float* res_out = (float*)d_out + (size_t)BB * 3 * N4; // (B,64,4N)

  // ---- workspace carve ----
  char* p = (char*)d_ws;
  auto alloc = [&](size_t bytes) -> void* {
    void* r = (void*)p;
    p += (bytes + 255) & ~(size_t)255;
    return r;
  };
  int*       idx    = (int*)alloc((size_t)BB * KK * NN * 4);
  _Float16*  featT  = (_Float16*)alloc((size_t)BB * CIN * NN * 2);
  _Float16*  bt1_h  = (_Float16*)alloc((size_t)DD * CIN * 2);
  _Float16*  bt2_h  = (_Float16*)alloc((size_t)DD * DD * 2);
  _Float16*  bts_h  = (_Float16*)alloc((size_t)DD * CIN * 2);
  _Float16*  wq_h   = (_Float16*)alloc((size_t)DD * DD * 2);
  _Float16*  wk_h   = (_Float16*)alloc((size_t)DD * DD * 2);
  _Float16*  wv_h   = (_Float16*)alloc((size_t)DD * DD * 2);
  _Float16*  fd2_h  = (_Float16*)alloc((size_t)DD * DD * 2);
  _Float16*  fg1_h  = (_Float16*)alloc((size_t)4 * DD * DD * 2);
  _Float16*  fg2_h  = (_Float16*)alloc((size_t)4 * DD * DD * 2);
  _Float16*  m41_h  = (_Float16*)alloc((size_t)DD * DD * 2);
  _Float16*  atT_h  = (_Float16*)alloc((size_t)4 * DD * DD * 2);
  float*     gsA    = (float*)alloc((size_t)DD * 4);
  float*     gsG    = (float*)alloc((size_t)4 * DD * 4);
  float*     xo     = (float*)alloc((size_t)BB * DD * NN * 4);
  float*     qo     = (float*)alloc((size_t)BB * DD * NN * 4);
  float*     ko     = (float*)alloc((size_t)BB * DD * NN * 4);
  float*     vo     = (float*)alloc((size_t)BB * DD * NN * 4);
  float*     value  = (float*)alloc((size_t)BB * DD * KN * 4);
  _Float16*  sT     = (_Float16*)alloc((size_t)BB * DD * KN * 2);
  _Float16*  attnT  = (_Float16*)alloc((size_t)BB * DD * KN * 2);
  _Float16*  resT   = (_Float16*)alloc((size_t)BB * DD * N4 * 2);

  // ---- converts ----
  auto cvt = [&](const float* s, _Float16* d, int n) {
    int g = (n + 255) / 256; if (g > 1024) g = 1024;
    cvt_f32_f16_kernel<<<g, 256, 0, stream>>>(s, d, n);
  };
  feat_transpose_kernel<<<1024, 256, 0, stream>>>(feature, featT);
  cvt(bt1_w, bt1_h, DD * CIN);
  cvt(bt2_w, bt2_h, DD * DD);
  cvt(bts_w, bts_h, DD * CIN);
  cvt(q_w, wq_h, DD * DD);
  cvt(k_w, wk_h, DD * DD);
  cvt(v_w, wv_h, DD * DD);
  cvt(fd2_w, fd2_h, DD * DD);
  cvt(fg1_w, fg1_h, 4 * DD * DD);
  cvt(fg2_w, fg2_h, 4 * DD * DD);
  cvt(m41_w, m41_h, DD * DD);
  prep_misc_kernel<<<1, 256, 0, stream>>>(at_w, fd_g, fg_g, atT_h, gsA, gsG);

  // ---- pipeline ----
  knn_kernel<<<(BB * NN + 255) / 256, 256, 0, stream>>>(xyz, idx);

  backbone_kernel<<<BB * (NN / 16), 128, 0, stream>>>(
      featT, bt1_h, bt1_b, bt2_h, bt2_b, bts_h, bts_b,
      wq_h, q_b, wk_h, k_b, wv_h, v_b, xo, qo, ko, vo);

  posenc_kernel<<<BB * (KN / 16), 128, 0, stream>>>(
      xyz, idx, fd1_w, fd1_b, gsA, fd_bb, fd2_h, fd2_b,
      qo, ko, vo, value, sT);

  attnmlp_kernel<<<BB * (KN / 16), 128, 0, stream>>>(
      sT, fg1_h, fg1_b, gsG, fg_bb, fg2_h, fg2_b, attnT);

  fuse_kernel<<<BB * NN, 128, 0, stream>>>(
      attnT, atT_h, at_b, value, xo, res_out, resT);

  completion_kernel<<<BB * (N4 / 16), 128, 0, stream>>>(
      resT, m41_h, m41_b, m42_w, m42_b, comp);
}
